// TransformerBlock_49443663511803
// MI455X (gfx1250) — compile-verified, matrix operand order checked by reading, not run on hardware
//
#include <hip/hip_runtime.h>
#include <hip/hip_bf16.h>
#include <math.h>

// ---------------------------------------------------------------------------
// Transformer block on CDNA5 (gfx1250), wave32, bf16 WMMA w/ f32 accumulation.
// Pipeline:
//   1) convert x -> bf16; transpose+convert weights -> bf16 [N,K]
//   2) WMMA GEMMs: Q (f32), K (f32), V (bf16, head-transposed [b,h,d,s])
//   3) flash attention (scores = K.Q^T per the reference's einsum), online
//      softmax, P.V via WMMA, LDS transpose for the P fragment
//   4) x1 = LN(attn + Q) -> f32 + bf16
//   5) h = relu(x1 @ W1 + c1) (WMMA, bf16 out)
//   6) f = h @ W2 + c2 (WMMA, f32 out)
//   7) out = LN(LN(f) + x1)
// ---------------------------------------------------------------------------

typedef __attribute__((ext_vector_type(16))) __bf16 v16bf;
typedef __attribute__((ext_vector_type(8)))  float  v8f;

#define S_LEN  2048
#define E_DIM  512
#define V_DIM  512
#define F_DIM  2048
#define B_SZ   4
#define H_CNT  8
#define D_HEAD 64
#define M_ROWS (B_SZ * S_LEN)

__device__ __forceinline__ unsigned short f2bf(float f) {
  union { float f; unsigned u; } a; a.f = f;
  unsigned u = a.u;
  return (unsigned short)((u + 0x7FFFu + ((u >> 16) & 1u)) >> 16); // RNE
}

// A/B fragment loader for v_wmma_f32_16x16x32_bf16 from row-major bf16 data.
// Layout (ISA 7.12.2, 16-bit A 16x32): lane L holds row (L&15); lanes 0-15
// carry K = [k0, k0+8) and [k0+16, k0+24); lanes 16-31 the +8 halves.
// The same layout serves the B operand when loading from a [N,K] (transposed)
// matrix: lane holds column n = (L&15) of B.
__device__ __forceinline__ v16bf load_frag_bf16(const unsigned short* base, int ld,
                                                int row0, int k0, int lane) {
  const int half = lane >> 4;
  const unsigned short* p = base + (size_t)(row0 + (lane & 15)) * ld + k0 + 8 * half;
  union { uint4 q[2]; v16bf v; } u;
  u.q[0] = *(const uint4*)p;
  u.q[1] = *(const uint4*)(p + 16);
  return u.v;
}

// Same fragment, sourced from fp32 with on-the-fly bf16 conversion.
__device__ __forceinline__ v16bf load_frag_f32(const float* base, int ld,
                                               int row0, int k0, int lane) {
  const int half = lane >> 4;
  const float* p = base + (size_t)(row0 + (lane & 15)) * ld + k0 + 8 * half;
  union { unsigned short s[16]; v16bf v; } u;
#pragma unroll
  for (int e = 0; e < 8; ++e) {
    u.s[e]     = f2bf(p[e]);
    u.s[8 + e] = f2bf(p[16 + e]);
  }
  return u.v;
}

__device__ __forceinline__ v8f wmma_bf16(v16bf a, v16bf b, v8f c) {
  return __builtin_amdgcn_wmma_f32_16x16x32_bf16(false, a, false, b, (short)0, c,
                                                 false, false);
}

// ---------------------------------------------------------------------------
// Elementwise fp32 -> bf16
__global__ void __launch_bounds__(256)
cvt_bf16_kernel(const float* __restrict__ in, unsigned short* __restrict__ out, int n) {
  int i = blockIdx.x * 256 + threadIdx.x;
  if (i < n) out[i] = f2bf(in[i]);
}

// [Kd,Nd] fp32 -> [Nd,Kd] bf16
__global__ void __launch_bounds__(256)
transpose_bf16_kernel(const float* __restrict__ in, unsigned short* __restrict__ out,
                      int Kd, int Nd) {
  int i = blockIdx.x * 256 + threadIdx.x;
  if (i < Kd * Nd) {
    int k = i / Nd, n = i % Nd;
    out[(size_t)n * Kd + k] = f2bf(in[i]);
  }
}

// ---------------------------------------------------------------------------
// C[M,N] = A[M,K] @ Bt[N,K]^T (+bias)(+relu).  Block = 8 waves, each wave owns
// a 16x64 tile; grid = (M/128, N/64).
// mode 0: f32 out; mode 1: bf16 out; mode 2: bf16 out transposed per-head
//         to Vt[b][h][d][s] (for the attention's P.V B-operand).
__global__ void __launch_bounds__(256)
gemm_bf16_kernel(const unsigned short* __restrict__ A,
                 const unsigned short* __restrict__ Bt,
                 const float* __restrict__ bias,
                 float* __restrict__ Cf, unsigned short* __restrict__ Cb,
                 int M, int N, int K, int mode, int relu) {
  const int lane = threadIdx.x & 31;
  const int wave = threadIdx.x >> 5;
  const int m0 = blockIdx.x * 128 + wave * 16;
  const int n0 = blockIdx.y * 64;

  v8f acc0 = {}, acc1 = {}, acc2 = {}, acc3 = {};
  for (int k0 = 0; k0 < K; k0 += 32) {
    v16bf a  = load_frag_bf16(A,  K, m0,      k0, lane);
    v16bf b0 = load_frag_bf16(Bt, K, n0 +  0, k0, lane);
    v16bf b1 = load_frag_bf16(Bt, K, n0 + 16, k0, lane);
    v16bf b2 = load_frag_bf16(Bt, K, n0 + 32, k0, lane);
    v16bf b3 = load_frag_bf16(Bt, K, n0 + 48, k0, lane);
    acc0 = wmma_bf16(a, b0, acc0);
    acc1 = wmma_bf16(a, b1, acc1);
    acc2 = wmma_bf16(a, b2, acc2);
    acc3 = wmma_bf16(a, b3, acc3);
  }

  const int col = lane & 15;
  const int h2  = lane >> 4;
  v8f accs[4] = {acc0, acc1, acc2, acc3};
#pragma unroll
  for (int t = 0; t < 4; ++t) {
    int n = n0 + 16 * t + col;
    float bv = bias ? bias[n] : 0.0f;
#pragma unroll
    for (int e = 0; e < 8; ++e) {
      int m = m0 + e + 8 * h2;                 // C layout: VGPR e -> row e (+8)
      float v = accs[t][e] + bv;
      if (relu) v = fmaxf(v, 0.0f);
      if (mode == 0) {
        Cf[(size_t)m * N + n] = v;
      } else if (mode == 1) {
        Cb[(size_t)m * N + n] = f2bf(v);
      } else {
        int bb = m / S_LEN, s = m % S_LEN;
        int hh = n >> 6,    d = n & 63;
        Cb[(((size_t)bb * H_CNT + hh) * D_HEAD + d) * S_LEN + s] = f2bf(v);
      }
    }
  }
}

// ---------------------------------------------------------------------------
// Flash attention. Reference einsum gives corr[b,h,i,j] = K[b,i,h,:].Q[b,j,h,:],
// softmax over j, out[i,d] = sum_j P[i,j] V[j,d].
// Block = 8 waves; wave owns 16 output rows i; streams keys in tiles of 32.
__global__ void __launch_bounds__(256)
flash_attn_kernel(const float* __restrict__ Kf, const float* __restrict__ Qf,
                  const unsigned short* __restrict__ Vt, const int* __restrict__ mask,
                  float* __restrict__ Out, float scale) {
  __shared__ __align__(16) unsigned short ptile[8][16][32];

  const int lane = threadIdx.x & 31;
  const int wave = threadIdx.x >> 5;
  const int h2   = lane >> 4;
  const int col  = lane & 15;
  const int b = blockIdx.z, h = blockIdx.y;
  const int i0 = blockIdx.x * 128 + wave * 16;

  const float* Kb_ = Kf + (size_t)b * S_LEN * E_DIM + h * D_HEAD;
  const float* Qb_ = Qf + (size_t)b * S_LEN * E_DIM + h * D_HEAD;
  const unsigned short* Vb_ = Vt + ((size_t)b * H_CNT + h) * D_HEAD * S_LEN;

  // A operand: this wave's 16 K-rows over d = 0..63 (reused for all key tiles)
  v16bf aK0 = load_frag_f32(Kb_, E_DIM, i0, 0,  lane);
  v16bf aK1 = load_frag_f32(Kb_, E_DIM, i0, 32, lane);

  v8f O0 = {}, O1 = {}, O2 = {}, O3 = {};
  float mrow[8], lrow[8];
#pragma unroll
  for (int e = 0; e < 8; ++e) { mrow[e] = -3.0e38f; lrow[e] = 0.0f; }

  const float mval = -1.0e20f * scale;

  for (int j0 = 0; j0 < S_LEN; j0 += 32) {
    // scores tile 16x32: two 16x16 WMMA fragments, K over d (2 steps of 32)
    v16bf bq00 = load_frag_f32(Qb_, E_DIM, j0 +  0, 0,  lane);
    v16bf bq01 = load_frag_f32(Qb_, E_DIM, j0 +  0, 32, lane);
    v16bf bq10 = load_frag_f32(Qb_, E_DIM, j0 + 16, 0,  lane);
    v16bf bq11 = load_frag_f32(Qb_, E_DIM, j0 + 16, 32, lane);
    v8f s0 = {}, s1 = {};
    s0 = wmma_bf16(aK0, bq00, s0); s0 = wmma_bf16(aK1, bq01, s0);
    s1 = wmma_bf16(aK0, bq10, s1); s1 = wmma_bf16(aK1, bq11, s1);

    float p0[8], p1[8];
#pragma unroll
    for (int e = 0; e < 8; ++e) {
      size_t i = (size_t)(i0 + e + 8 * h2);
      int j = j0 + col;
      float v0 = mask[i * S_LEN + j]      ? s0[e] * scale : mval;
      float v1 = mask[i * S_LEN + j + 16] ? s1[e] * scale : mval;
      // row-wise max/sum: reduce across the 16 lanes of this half
      float mx = fmaxf(v0, v1);
#pragma unroll
      for (int off = 1; off < 16; off <<= 1) mx = fmaxf(mx, __shfl_xor(mx, off, 32));
      float mnew  = fmaxf(mrow[e], mx);
      float alpha = __expf(mrow[e] - mnew);
      mrow[e] = mnew;
      float e0 = __expf(v0 - mnew), e1 = __expf(v1 - mnew);
      float rs = e0 + e1;
#pragma unroll
      for (int off = 1; off < 16; off <<= 1) rs += __shfl_xor(rs, off, 32);
      lrow[e] = lrow[e] * alpha + rs;
      p0[e] = e0; p1[e] = e1;
      O0[e] *= alpha; O1[e] *= alpha; O2[e] *= alpha; O3[e] *= alpha;
    }

    // transpose P (C-layout) -> A-layout via per-wave LDS tile
#pragma unroll
    for (int e = 0; e < 8; ++e) {
      ptile[wave][e + 8 * h2][col]      = f2bf(p0[e]);
      ptile[wave][e + 8 * h2][col + 16] = f2bf(p1[e]);
    }
    __builtin_amdgcn_wave_barrier();   // LDS is in-order within a wave
    union { uint4 q[2]; v16bf v; } up;
    const unsigned short* prow = &ptile[wave][col][0];
    up.q[0] = *(const uint4*)(prow + 8 * h2);
    up.q[1] = *(const uint4*)(prow + 16 + 8 * h2);
    __builtin_amdgcn_wave_barrier();
    v16bf pa = up.v;

    // O += P . V  (B operand from head-transposed Vt rows = d, cols = j)
    v16bf bv0 = load_frag_bf16(Vb_, S_LEN, 0,  j0, lane);
    v16bf bv1 = load_frag_bf16(Vb_, S_LEN, 16, j0, lane);
    v16bf bv2 = load_frag_bf16(Vb_, S_LEN, 32, j0, lane);
    v16bf bv3 = load_frag_bf16(Vb_, S_LEN, 48, j0, lane);
    O0 = wmma_bf16(pa, bv0, O0);
    O1 = wmma_bf16(pa, bv1, O1);
    O2 = wmma_bf16(pa, bv2, O2);
    O3 = wmma_bf16(pa, bv3, O3);
  }

#pragma unroll
  for (int e = 0; e < 8; ++e) {
    float inv = 1.0f / lrow[e];
    size_t i = (size_t)(i0 + e + 8 * h2);
    float* orow = Out + ((size_t)b * S_LEN + i) * E_DIM + h * D_HEAD + col;
    orow[0]  = O0[e] * inv;
    orow[16] = O1[e] * inv;
    orow[32] = O2[e] * inv;
    orow[48] = O3[e] * inv;
  }
}

// ---------------------------------------------------------------------------
// x1 = LN(src + res) * g + b  -> f32 (+ bf16 copy). One block per row.
__global__ void __launch_bounds__(256)
add_ln_kernel(const float* __restrict__ src, const float* __restrict__ res,
              const float* __restrict__ g, const float* __restrict__ bb,
              float* __restrict__ outf, unsigned short* __restrict__ outb) {
  __shared__ float r1[256], r2[256];
  const int row = blockIdx.x;
  const int t = threadIdx.x;
  const float* s = src + (size_t)row * E_DIM;
  const float* r = res + (size_t)row * E_DIM;
  float x0 = s[t]       + r[t];
  float x1 = s[t + 256] + r[t + 256];
  r1[t] = x0 + x1;
  r2[t] = x0 * x0 + x1 * x1;
  __syncthreads();
  for (int o = 128; o > 0; o >>= 1) {
    if (t < o) { r1[t] += r1[t + o]; r2[t] += r2[t + o]; }
    __syncthreads();
  }
  float mean = r1[0] * (1.0f / E_DIM);
  float var  = r2[0] * (1.0f / E_DIM) - mean * mean;
  float rstd = rsqrtf(var + 1e-5f);
  float y0 = (x0 - mean) * rstd * g[t]       + bb[t];
  float y1 = (x1 - mean) * rstd * g[t + 256] + bb[t + 256];
  float* of = outf + (size_t)row * E_DIM;
  of[t] = y0; of[t + 256] = y1;
  unsigned short* ob = outb + (size_t)row * E_DIM;
  ob[t] = f2bf(y0); ob[t + 256] = f2bf(y1);
}

// out = LN( LN(f)*g3+b3 + x1 ) * g2 + b2.  One block per row.
__global__ void __launch_bounds__(256)
double_ln_kernel(const float* __restrict__ f, const float* __restrict__ x1,
                 const float* __restrict__ g3, const float* __restrict__ b3,
                 const float* __restrict__ g2, const float* __restrict__ b2,
                 float* __restrict__ out) {
  __shared__ float r1[256], r2[256];
  const int row = blockIdx.x;
  const int t = threadIdx.x;
  const float* fr = f  + (size_t)row * E_DIM;
  const float* xr = x1 + (size_t)row * E_DIM;
  float a0 = fr[t], a1 = fr[t + 256];
  r1[t] = a0 + a1; r2[t] = a0 * a0 + a1 * a1;
  __syncthreads();
  for (int o = 128; o > 0; o >>= 1) {
    if (t < o) { r1[t] += r1[t + o]; r2[t] += r2[t + o]; }
    __syncthreads();
  }
  float mean = r1[0] * (1.0f / E_DIM);
  float var  = r2[0] * (1.0f / E_DIM) - mean * mean;
  float rstd = rsqrtf(var + 1e-5f);
  float u0 = (a0 - mean) * rstd * g3[t]       + b3[t]       + xr[t];
  float u1 = (a1 - mean) * rstd * g3[t + 256] + b3[t + 256] + xr[t + 256];
  __syncthreads();
  r1[t] = u0 + u1; r2[t] = u0 * u0 + u1 * u1;
  __syncthreads();
  for (int o = 128; o > 0; o >>= 1) {
    if (t < o) { r1[t] += r1[t + o]; r2[t] += r2[t + o]; }
    __syncthreads();
  }
  mean = r1[0] * (1.0f / E_DIM);
  var  = r2[0] * (1.0f / E_DIM) - mean * mean;
  rstd = rsqrtf(var + 1e-5f);
  float* orow = out + (size_t)row * E_DIM;
  orow[t]       = (u0 - mean) * rstd * g2[t]       + b2[t];
  orow[t + 256] = (u1 - mean) * rstd * g2[t + 256] + b2[t + 256];
}

// ---------------------------------------------------------------------------
extern "C" void kernel_launch(void* const* d_in, const int* in_sizes, int n_in,
                              void* d_out, int out_size, void* d_ws, size_t ws_size,
                              hipStream_t stream) {
  const float* x  = (const float*)d_in[0];
  const int* mask = (const int*)d_in[1];
  const float* Wv = (const float*)d_in[2];
  const float* Wk = (const float*)d_in[3];
  const float* Wq = (const float*)d_in[4];
  const float* g1 = (const float*)d_in[5];
  const float* b1 = (const float*)d_in[6];
  const float* W1 = (const float*)d_in[7];
  const float* c1 = (const float*)d_in[8];
  const float* W2 = (const float*)d_in[9];
  const float* c2 = (const float*)d_in[10];
  const float* g3 = (const float*)d_in[11];
  const float* b3 = (const float*)d_in[12];
  const float* g2 = (const float*)d_in[13];
  const float* b2 = (const float*)d_in[14];

  const size_t MB = 1ull << 20;
  unsigned char* w = (unsigned char*)d_ws;
  // live regions (peak ~96 MB, with reuse):
  unsigned short* xb  = (unsigned short*)(w +  0 * MB); //  8 MB  x bf16
  unsigned short* Wqt = (unsigned short*)(w +  8 * MB); //  .5 MB
  unsigned short* Wkt = (unsigned short*)(w +  9 * MB); //  .5 MB
  unsigned short* Wvt = (unsigned short*)(w + 10 * MB); //  .5 MB
  unsigned short* W1t = (unsigned short*)(w + 11 * MB); //  2 MB
  unsigned short* W2t = (unsigned short*)(w + 13 * MB); //  2 MB
  float*          Kf  = (float*)         (w + 16 * MB); // 16 MB  keys f32
  unsigned short* Vt  = (unsigned short*)(w + 32 * MB); //  8 MB  V^T bf16
  float*          Qf  = (float*)         (w + 40 * MB); // 16 MB  queries f32
  float*          attn= (float*)         (w + 56 * MB); // 16 MB
  float*          x1f = (float*)         (w + 72 * MB); // 16 MB
  unsigned short* x1b = (unsigned short*)(w + 88 * MB); //  8 MB
  unsigned short* hb  = (unsigned short*)(w + 16 * MB); // 32 MB (reuses Kf/Vt/Qf)
  float*          ff  = (float*)         (w + 48 * MB); // 16 MB (reuses Qf/attn)

  dim3 blk(256);

  // 1) conversions / weight transposes
  cvt_bf16_kernel<<<(M_ROWS * V_DIM) / 256, blk, 0, stream>>>(x, xb, M_ROWS * V_DIM);
  transpose_bf16_kernel<<<(V_DIM * E_DIM) / 256, blk, 0, stream>>>(Wq, Wqt, V_DIM, E_DIM);
  transpose_bf16_kernel<<<(V_DIM * E_DIM) / 256, blk, 0, stream>>>(Wk, Wkt, V_DIM, E_DIM);
  transpose_bf16_kernel<<<(V_DIM * E_DIM) / 256, blk, 0, stream>>>(Wv, Wvt, V_DIM, E_DIM);
  transpose_bf16_kernel<<<(E_DIM * F_DIM) / 256, blk, 0, stream>>>(W1, W1t, E_DIM, F_DIM);
  transpose_bf16_kernel<<<(F_DIM * E_DIM) / 256, blk, 0, stream>>>(W2, W2t, F_DIM, E_DIM);

  // 2) QKV projections
  dim3 gq(M_ROWS / 128, E_DIM / 64);
  gemm_bf16_kernel<<<gq, blk, 0, stream>>>(xb, Wqt, nullptr, Qf, nullptr,
                                           M_ROWS, E_DIM, V_DIM, 0, 0);
  gemm_bf16_kernel<<<gq, blk, 0, stream>>>(xb, Wkt, nullptr, Kf, nullptr,
                                           M_ROWS, E_DIM, V_DIM, 0, 0);
  gemm_bf16_kernel<<<gq, blk, 0, stream>>>(xb, Wvt, nullptr, nullptr, Vt,
                                           M_ROWS, E_DIM, V_DIM, 2, 0);

  // 3) fused attention
  dim3 ga(S_LEN / 128, H_CNT, B_SZ);
  const float scale = 1.0f / sqrtf((float)E_DIM);
  flash_attn_kernel<<<ga, blk, 0, stream>>>(Kf, Qf, Vt, mask, attn, scale);

  // 4) x1 = LN(attn + queries)
  add_ln_kernel<<<M_ROWS, blk, 0, stream>>>(attn, Qf, g1, b1, x1f, x1b);

  // 5) h = relu(x1 @ W1 + c1)
  dim3 gf1(M_ROWS / 128, F_DIM / 64);
  gemm_bf16_kernel<<<gf1, blk, 0, stream>>>(x1b, W1t, c1, nullptr, hb,
                                            M_ROWS, F_DIM, E_DIM, 1, 1);

  // 6) f = h @ W2 + c2
  dim3 gf2(M_ROWS / 128, E_DIM / 64);
  gemm_bf16_kernel<<<gf2, blk, 0, stream>>>(hb, W2t, c2, ff, nullptr,
                                            M_ROWS, E_DIM, F_DIM, 0, 0);

  // 7) out = LN(LN(f) + x1)
  double_ln_kernel<<<M_ROWS, blk, 0, stream>>>(ff, x1f, g3, b3, g2, b2,
                                               (float*)d_out);
}
